// _IPEXVarlenScaledDotProductCPU_54477365182767
// MI455X (gfx1250) — matrix operand order repaired and checked by reading, not compiled, not run on hardware
//
#include <hip/hip_runtime.h>

// Varlen causal flash-attention fwd for MI455X (gfx1250, wave32).
// One wave per (batch, q-head, 32-row Q block). S computed TRANSPOSED
// (S^T = K * Q^T): each lane owns a q row (in-lane softmax), and P^T in C/D
// layout is lane-identical to the A-layout needed for P*V (no LDS transpose).
// Pre-pass converts K->f16 (row major) and V->f16 TRANSPOSED [hk][d][tok] in
// the workspace, so inner-loop operands are pure global_load_b128 of f16 --
// no per-iteration f32->f16 conversion (KV data is reused ~256x).

typedef __attribute__((ext_vector_type(16))) _Float16 v16h;
typedef __attribute__((ext_vector_type(8)))  _Float16 v8h;
typedef __attribute__((ext_vector_type(8)))  float    v8f;
typedef __attribute__((ext_vector_type(4)))  float    v4f;

#define HQ   16
#define HKV  4
#define NREP 4
#define DH   64
#define KVSTRIDE (HKV * DH)            // elements between consecutive kv tokens
#define WSPAD 32768                    // safe overread pad after vt region

#if __has_builtin(__builtin_amdgcn_exp2f)
#define EXP2F(x) __builtin_amdgcn_exp2f(x)
#else
#define EXP2F(x) exp2f(x)
#endif

// softmax scale folded with log2(e); applied to Q at load time.
#define SSCALE 0.18033688011112042f

static __device__ __forceinline__ v16h cat8(v8h a, v8h b)
{
    return __builtin_shufflevector(a, b, 0,1,2,3,4,5,6,7,8,9,10,11,12,13,14,15);
}

// ---------------- shared online-softmax / P prep (S^T orientation) ----------
template <bool MASKED>
__device__ __forceinline__ void softmax_update(
    int kb, int q0, int lenk, int nl, int hf,
    v8f (&st)[2][2], v16h (&pa)[2],
    v8f (&oacc)[2][4], float (&mrow)[2], float (&lpart)[2])
{
    #pragma unroll
    for (int qt = 0; qt < 2; ++qt) {
        const int qrow = q0 + qt * 16 + nl;       // this lane's q row
        if (MASKED) {
            #pragma unroll
            for (int nt = 0; nt < 2; ++nt)
                #pragma unroll
                for (int r = 0; r < 8; ++r) {
                    int kv = kb + nt * 16 + 8 * hf + r;
                    bool ok = (kv <= qrow) && (kv < lenk);
                    st[qt][nt][r] = ok ? st[qt][nt][r] : -1e30f;
                }
        }
        // in-lane max over 16 kv values, then combine the two kv halves
        float cm = fmaxf(st[qt][0][0], st[qt][1][0]);
        #pragma unroll
        for (int r = 1; r < 8; ++r)
            cm = fmaxf(cm, fmaxf(st[qt][0][r], st[qt][1][r]));
        cm = fmaxf(cm, __shfl_xor(cm, 16, 32));
        float mn = fmaxf(mrow[qt], cm);
        float al = EXP2F(mrow[qt] - mn);
        mrow[qt] = mn;

        float ssum = 0.0f;
        #pragma unroll
        for (int nt = 0; nt < 2; ++nt)
            #pragma unroll
            for (int r = 0; r < 8; ++r) {
                float p = EXP2F(st[qt][nt][r] - mn);
                st[qt][nt][r] = p;
                ssum += p;
            }
        lpart[qt] = lpart[qt] * al + ssum;

        // O rows m=8*hf+r need alpha of q row (qt*16 + 8*hf + r)
        #pragma unroll
        for (int r = 0; r < 8; ++r) {
            float alr = __shfl(al, 8 * hf + r, 32);
            #pragma unroll
            for (int t = 0; t < 4; ++t) oacc[qt][t][r] *= alr;
        }
        // P^T D-layout == A-layout this lane needs for P*V
        #pragma unroll
        for (int e = 0; e < 8; ++e) {
            pa[qt][e]     = (_Float16)st[qt][0][e];
            pa[qt][e + 8] = (_Float16)st[qt][1][e];
        }
    }
}

// ---------------- fast path: f16 KV from workspace ---------------------------
template <bool MASKED>
__device__ __forceinline__ void kv_step_f16(
    int kb, int q0, int lenk, int nl, int hf,
    const _Float16* __restrict__ kp0,   // f16 K base for (batch, kv-head)
    const _Float16* __restrict__ vt0,   // f16 V^T base: + d*vstride + tok
    size_t vstride,
    const v16h (&qb)[2][2],
    v8f (&oacc)[2][4], float (&mrow)[2], float (&lpart)[2])
{
    v8f st[2][2];
    #pragma unroll
    for (int nt = 0; nt < 2; ++nt) {
        const _Float16* kp = kp0 + (size_t)(kb + nt * 16 + nl) * KVSTRIDE;
        const v8h* k8 = (const v8h*)kp;            // 8-half groups, 16B aligned
        v16h ka0 = cat8(k8[hf],     k8[2 + hf]);   // d in [8hf,+8) U [16+8hf,+8)
        v16h ka1 = cat8(k8[4 + hf], k8[6 + hf]);   // +32
        #pragma unroll
        for (int qt = 0; qt < 2; ++qt) {
            v8f acc = {};
            acc = __builtin_amdgcn_wmma_f32_16x16x32_f16(false, ka0, false, qb[qt][0],
                                                         (short)0, acc, false, false);
            acc = __builtin_amdgcn_wmma_f32_16x16x32_f16(false, ka1, false, qb[qt][1],
                                                         (short)0, acc, false, false);
            st[qt][nt] = acc;
        }
    }

    v16h pa[2];
    softmax_update<MASKED>(kb, q0, lenk, nl, hf, st, pa, oacc, mrow, lpart);

    #pragma unroll
    for (int t = 0; t < 4; ++t) {
        // V^T: tokens [8hf,+8) U [16+8hf,+8) for column d = 16t+nl (masked rows have p==0;
        // overreads land on finite converted data / zeroed pad)
        const v8h* v8 = (const v8h*)(vt0 + (size_t)(16 * t + nl) * vstride + kb + 8 * hf);
        v16h bv = cat8(v8[0], v8[2]);
        #pragma unroll
        for (int qt = 0; qt < 2; ++qt)
            oacc[qt][t] = __builtin_amdgcn_wmma_f32_16x16x32_f16(false, pa[qt], false, bv,
                                                                 (short)0, oacc[qt][t], false, false);
    }
}

// ---------------- fallback path: direct fp32 KV ------------------------------
template <bool MASKED>
__device__ __forceinline__ void kv_step_f32(
    int kb, int q0, int lenk, int nl, int hf,
    const float* __restrict__ kp0, const float* __restrict__ vp0,
    const v16h (&qb)[2][2],
    v8f (&oacc)[2][4], float (&mrow)[2], float (&lpart)[2])
{
    v8f st[2][2];
    #pragma unroll
    for (int nt = 0; nt < 2; ++nt) {
        int kvr = kb + nt * 16 + nl;
        if (MASKED) { if (kvr >= lenk) kvr = lenk - 1; }
        const float* kp = kp0 + (size_t)kvr * KVSTRIDE;
        v16h ka0, ka1;
        #pragma unroll
        for (int e = 0; e < 8; ++e) {
            ka0[e]     = (_Float16)kp[     8 * hf + e];
            ka0[e + 8] = (_Float16)kp[16 + 8 * hf + e];
            ka1[e]     = (_Float16)kp[32 + 8 * hf + e];
            ka1[e + 8] = (_Float16)kp[48 + 8 * hf + e];
        }
        #pragma unroll
        for (int qt = 0; qt < 2; ++qt) {
            v8f acc = {};
            acc = __builtin_amdgcn_wmma_f32_16x16x32_f16(false, ka0, false, qb[qt][0],
                                                         (short)0, acc, false, false);
            acc = __builtin_amdgcn_wmma_f32_16x16x32_f16(false, ka1, false, qb[qt][1],
                                                         (short)0, acc, false, false);
            st[qt][nt] = acc;
        }
    }

    v16h pa[2];
    softmax_update<MASKED>(kb, q0, lenk, nl, hf, st, pa, oacc, mrow, lpart);

    #pragma unroll
    for (int t = 0; t < 4; ++t) {
        v16h bv;
        #pragma unroll
        for (int e = 0; e < 16; ++e) {
            int kvr = kb + 8 * hf + (e & 7) + 16 * (e >> 3);
            if (MASKED) { if (kvr >= lenk) kvr = lenk - 1; }
            bv[e] = (_Float16)vp0[(size_t)kvr * KVSTRIDE + 16 * t + nl];
        }
        #pragma unroll
        for (int qt = 0; qt < 2; ++qt)
            oacc[qt][t] = __builtin_amdgcn_wmma_f32_16x16x32_f16(false, pa[qt], false, bv,
                                                                 (short)0, oacc[qt][t], false, false);
    }
}

// ---------------- pre-pass kernels -------------------------------------------
__global__ __launch_bounds__(256)
void cvt_k_f16(const float* __restrict__ src, _Float16* __restrict__ dst, int n8)
{
    int i = blockIdx.x * blockDim.x + threadIdx.x;
    if (i < n8) {
        const v4f* s = (const v4f*)src + (size_t)i * 2;
        v4f a = s[0], b = s[1];
        v8h o;
        #pragma unroll
        for (int e = 0; e < 4; ++e) { o[e] = (_Float16)a[e]; o[e + 4] = (_Float16)b[e]; }
        ((v8h*)dst)[i] = o;
    }
}

__global__ __launch_bounds__(256)
void cvt_vt_f16(const float* __restrict__ v, _Float16* __restrict__ vt, int total)
{
    __shared__ _Float16 tile[32][33];
    const int hk = blockIdx.z;
    const int d0 = blockIdx.y * 32;
    const int t0 = blockIdx.x * 32;
    const int tx = threadIdx.x, ty = threadIdx.y;   // 32 x 8
    #pragma unroll
    for (int i = 0; i < 4; ++i) {
        int tok = t0 + ty + i * 8;
        if (tok < total)
            tile[ty + i * 8][tx] = (_Float16)v[((size_t)tok * HKV + hk) * DH + d0 + tx];
    }
    __syncthreads();
    if (t0 + tx < total) {
        #pragma unroll
        for (int i = 0; i < 4; ++i) {
            int d = d0 + ty + i * 8;
            vt[((size_t)hk * DH + d) * total + t0 + tx] = tile[tx][ty + i * 8];
        }
    }
}

__global__ __launch_bounds__(256)
void zero_pad_f16(_Float16* __restrict__ p, int n)
{
    int i = blockIdx.x * blockDim.x + threadIdx.x;
    if (i < n) p[i] = (_Float16)0.0f;
}

// ---------------- main kernels -----------------------------------------------
template <bool F16KV>
__global__ __launch_bounds__(32)
void varlen_causal_attn(const float* __restrict__ qg,
                        const void*  __restrict__ kptr,
                        const void*  __restrict__ vptr,
                        const int*   __restrict__ cuq,
                        const int*   __restrict__ cuk,
                        float*       __restrict__ out,
                        int totalK)
{
    const int lane = threadIdx.x;
    const int nl = lane & 15;
    const int hf = lane >> 4;

    const int b = blockIdx.z, h = blockIdx.y;
    const int hk = h / NREP;                       // GQA repeat_interleave
    const int qs = cuq[b], lenq = cuq[b + 1] - qs;
    const int ks = cuk[b], lenk = cuk[b + 1] - ks;
    const int q0 = blockIdx.x * 32;
    if (q0 >= lenq) return;                        // block-uniform exit

    // Q^T B-operands, pre-scaled by (1/sqrt(D))*log2(e): d = 32c + 8*hf + (e&7) + 16*(e>>3)
    v16h qb[2][2];
    #pragma unroll
    for (int qt = 0; qt < 2; ++qt) {
        int qr = q0 + qt * 16 + nl; if (qr >= lenq) qr = lenq - 1;
        const float* qp = qg + ((size_t)(qs + qr) * HQ + h) * DH;
        #pragma unroll
        for (int c = 0; c < 2; ++c)
            #pragma unroll
            for (int e = 0; e < 16; ++e)
                qb[qt][c][e] = (_Float16)(qp[32 * c + 8 * hf + (e & 7) + 16 * (e >> 3)] * SSCALE);
    }

    v8f oacc[2][4];
    float mrow[2] = {-1e30f, -1e30f};
    float lpart[2] = {0.0f, 0.0f};
    #pragma unroll
    for (int qt = 0; qt < 2; ++qt)
        #pragma unroll
        for (int t = 0; t < 4; ++t) { v8f z = {}; oacc[qt][t] = z; }

    const int kend = min(lenk, q0 + 32);
    int fullEnd = ((q0 + 1) >> 5) << 5;            // chunks with kb+31 <= q0
    if (fullEnd > kend) fullEnd = kend;

    if (F16KV) {
        const _Float16* kp0 = (const _Float16*)kptr + ((size_t)ks * HKV + hk) * DH;
        const _Float16* vt0 = (const _Float16*)vptr + (size_t)hk * DH * totalK + ks;
        int kb = 0;
        for (; kb < fullEnd; kb += 32)
            kv_step_f16<false>(kb, q0, lenk, nl, hf, kp0, vt0, (size_t)totalK, qb, oacc, mrow, lpart);
        for (; kb < kend; kb += 32)
            kv_step_f16<true>(kb, q0, lenk, nl, hf, kp0, vt0, (size_t)totalK, qb, oacc, mrow, lpart);
    } else {
        const float* kp0 = (const float*)kptr + ((size_t)ks * HKV + hk) * DH;
        const float* vp0 = (const float*)vptr + ((size_t)ks * HKV + hk) * DH;
        int kb = 0;
        for (; kb < fullEnd; kb += 32)
            kv_step_f32<false>(kb, q0, lenk, nl, hf, kp0, vp0, qb, oacc, mrow, lpart);
        for (; kb < kend; kb += 32)
            kv_step_f32<true>(kb, q0, lenk, nl, hf, kp0, vp0, qb, oacc, mrow, lpart);
    }

    // finalize: combine half-sums, broadcast 1/l to accumulator rows, store
    #pragma unroll
    for (int qt = 0; qt < 2; ++qt) {
        float lt = lpart[qt] + __shfl_xor(lpart[qt], 16, 32);
        float inv = 1.0f / lt;
        #pragma unroll
        for (int r = 0; r < 8; ++r) {
            float invr = __shfl(inv, 8 * hf + r, 32);    // before divergence
            int qrow = q0 + qt * 16 + 8 * hf + r;
            if (qrow < lenq) {
                float* op = out + ((size_t)(qs + qrow) * HQ + h) * DH + nl;
                #pragma unroll
                for (int t = 0; t < 4; ++t)
                    op[16 * t] = oacc[qt][t][r] * invr;
            }
        }
    }
}

extern "C" void kernel_launch(void* const* d_in, const int* in_sizes, int n_in,
                              void* d_out, int out_size, void* d_ws, size_t ws_size,
                              hipStream_t stream)
{
    const float* q   = (const float*)d_in[0];
    const float* k   = (const float*)d_in[1];
    const float* v   = (const float*)d_in[2];
    const int*   cuq = (const int*)d_in[3];
    const int*   cuk = (const int*)d_in[4];
    float*       out = (float*)d_out;

    const int B      = in_sizes[3] - 1;
    const int totalK = in_sizes[1] / KVSTRIDE;
    const size_t kelems = (size_t)in_sizes[1];
    const size_t velems = (size_t)in_sizes[2];
    const size_t need   = (kelems + velems) * sizeof(_Float16) + WSPAD;

    dim3 grid(1024 / 32, HQ, B);                 // max_seqlen=1024; extras exit

    if (ws_size >= need) {
        _Float16* kf = (_Float16*)d_ws;
        _Float16* vt = kf + kelems;
        _Float16* pad = vt + velems;

        int n8 = (int)(kelems / 8);
        cvt_k_f16<<<(n8 + 255) / 256, 256, 0, stream>>>(k, kf, n8);
        dim3 tg((totalK + 31) / 32, DH / 32, HKV);
        cvt_vt_f16<<<tg, dim3(32, 8), 0, stream>>>(v, vt, totalK);
        int npad = WSPAD / 2;
        zero_pad_f16<<<(npad + 255) / 256, 256, 0, stream>>>(pad, npad);

        varlen_causal_attn<true><<<grid, 32, 0, stream>>>(q, kf, vt, cuq, cuk, out, totalK);
    } else {
        varlen_causal_attn<false><<<grid, 32, 0, stream>>>(q, k, v, cuq, cuk, out, totalK);
    }
}
